// PointConv2_76175539962314
// MI455X (gfx1250) — matrix operand order repaired and checked by reading, not compiled, not run on hardware
//
#include <hip/hip_runtime.h>
#include <stdint.h>

// PointConv-style fused MLP pipeline for MI455X (gfx1250, wave32, WMMA).
// B=8, C=64, N=16384, K=8. GEMMs via v_wmma_f32_16x16x32_f16, register-blocked
// 4 m-tiles per wave-job; gathers via GLOBAL_LOAD_ASYNC_TO_LDS_B128
// (ASYNCcnt-tracked, s_wait_asynccnt fence).

typedef __attribute__((ext_vector_type(16))) _Float16 v16h;
typedef __attribute__((ext_vector_type(8)))  _Float16 v8h;
typedef __attribute__((ext_vector_type(8)))  float    v8f;
typedef __attribute__((ext_vector_type(4)))  int      v4i;

#define NB 8
#define CB 64
#define NP 16384
#define KN 8
#define TILE_N 64

// ---------------------------------------------------------------------------
// Async global->LDS copy helpers (CDNA5 ASYNCcnt path, with sync fallback)
// ---------------------------------------------------------------------------
#if __has_builtin(__builtin_amdgcn_global_load_async_to_lds_b128)
#define HAVE_ASYNC_LDS 1
#else
#define HAVE_ASYNC_LDS 0
#endif

static __device__ __forceinline__ void async_copy16(const _Float16* gsrc, _Float16* ldst)
{
#if HAVE_ASYNC_LDS
    auto g = (__attribute__((address_space(1))) v4i*)(uintptr_t)gsrc;
    auto l = (__attribute__((address_space(3))) v4i*)(uint32_t)(uintptr_t)ldst;
    __builtin_amdgcn_global_load_async_to_lds_b128(g, l, 0, 0);
#else
    *(v8h*)ldst = *(const v8h*)gsrc;
#endif
}

static __device__ __forceinline__ void async_copy32(const _Float16* gsrc, _Float16* ldst)
{
#if HAVE_ASYNC_LDS
    auto g = (__attribute__((address_space(1))) v4i*)(uintptr_t)gsrc;
    auto l = (__attribute__((address_space(3))) v4i*)(uint32_t)(uintptr_t)ldst;
    __builtin_amdgcn_global_load_async_to_lds_b128(g, l, 0, 0);
    __builtin_amdgcn_global_load_async_to_lds_b128(g, l, 16, 0);  // IOFFSET adds to both sides
#else
    ((v8h*)ldst)[0] = ((const v8h*)gsrc)[0];
    ((v8h*)ldst)[1] = ((const v8h*)gsrc)[1];
#endif
}

static __device__ __forceinline__ void async_wait0()
{
#if HAVE_ASYNC_LDS
#if __has_builtin(__builtin_amdgcn_s_wait_asynccnt)
    __builtin_amdgcn_s_wait_asynccnt(0);
#else
    asm volatile("s_wait_asynccnt 0" ::: "memory");
#endif
#endif
}

// ---------------------------------------------------------------------------
// WMMA helpers (layouts per CDNA5 ISA 7.12.2, wave32)
// ---------------------------------------------------------------------------

// Register-blocked accumulate: MB consecutive 16-row tiles of W against one
// 16-column tile of X. One B-fragment load per k-step feeds MB WMMAs.
// W: global f16 row-major [O][Ktot].  X: LDS f16 point-major [n][Ktot].
template<int MB>
static __device__ __forceinline__ void wmma_block(
    const _Float16* __restrict__ W, int Ktot, const _Float16* X,
    int mBase, int nBase, int lane, v8f acc[MB])
{
    const int selA = (lane & 16) ? 8 : 0;    // A: lanes>=16 hold K+8 run
    const int selB = (lane & 16) ? 8 : 0;    // B: lanes>=16 hold K+8 run
    const _Float16* arow = W + (size_t)(mBase + (lane & 15)) * Ktot + selA;
    const _Float16* bcol = X + (size_t)(nBase + (lane & 15)) * Ktot + selB;
    for (int k = 0; k < Ktot; k += 32) {
        union { v16h v; v8h h[2]; } Bm;
        Bm.h[0] = *(const v8h*)(bcol + k);
        Bm.h[1] = *(const v8h*)(bcol + k + 16);
#pragma unroll
        for (int m = 0; m < MB; ++m) {
            const _Float16* ar = arow + (size_t)m * 16 * Ktot + k;
            union { v16h v; v8h h[2]; } A;
            A.h[0] = *(const v8h*)ar;
            A.h[1] = *(const v8h*)(ar + 16);
            acc[m] = __builtin_amdgcn_wmma_f32_16x16x32_f16(
                false, A.v, false, Bm.v, (short)0, acc[m], false, false);
        }
    }
}

// D-fragment epilogue: bias + ReLU, store f16 into LDS [n][Odim].
static __device__ __forceinline__ void store_lds_relu(
    v8f acc, const float* __restrict__ bias,
    _Float16* Y, int Odim, int mBase, int nBase, int lane)
{
    const int n  = nBase + (lane & 15);
    const int mo = mBase + ((lane & 16) ? 8 : 0);
    _Float16* dst = Y + (size_t)n * Odim;
#pragma unroll
    for (int vi = 0; vi < 8; ++vi) {
        const int o = mo + vi;
        float val = acc[vi] + bias[o];
        dst[o] = (_Float16)(val > 0.f ? val : 0.f);
    }
}

// Full GEMM stage: Y_lds = relu(W @ X_lds + bias), M = Mtiles*16, N = 64.
template<int MB>
static __device__ __forceinline__ void gemm_stage_lds(
    const _Float16* __restrict__ W, const float* __restrict__ bias,
    int Mtiles, int Ktot, const _Float16* Xlds, _Float16* Ylds, int Odim,
    int w, int lane)
{
    const int jobs = (Mtiles / MB) * 4;
    for (int j = w; j < jobs; j += 8) {
        const int mb = j >> 2, nt = j & 3;
        v8f acc[MB];
#pragma unroll
        for (int m = 0; m < MB; ++m) acc[m] = (v8f){0.f,0.f,0.f,0.f,0.f,0.f,0.f,0.f};
        wmma_block<MB>(W, Ktot, Xlds, mb * MB * 16, nt * 16, lane, acc);
#pragma unroll
        for (int m = 0; m < MB; ++m)
            store_lds_relu(acc[m], bias, Ylds, Odim, (mb * MB + m) * 16, nt * 16, lane);
    }
}

// ---------------------------------------------------------------------------
// Stage 0: f32 -> f16 weight conversion
// ---------------------------------------------------------------------------
__global__ __launch_bounds__(256) void cvt_f32_f16(
    const float* __restrict__ src, _Float16* __restrict__ dst, int n)
{
    int i = blockIdx.x * 256 + threadIdx.x;
    if (i < n) dst[i] = (_Float16)src[i];
}

// ---------------------------------------------------------------------------
// Stage 1: feature [B,C,N] f32 -> featH [B,N,C] f16 (LDS-tiled transpose)
// ---------------------------------------------------------------------------
__global__ __launch_bounds__(256) void feat_transpose(
    const float* __restrict__ f, _Float16* __restrict__ featH, int Nn)
{
    __shared__ float tile[CB][TILE_N + 1];
    const int blocksPerBatch = Nn / TILE_N;
    const int bb = blockIdx.x / blocksPerBatch;
    const int n0 = (blockIdx.x % blocksPerBatch) * TILE_N;
    const int tid = threadIdx.x;
    for (int e = tid; e < CB * TILE_N; e += 256) {   // coalesced along n
        int c = e >> 6, nl = e & 63;
        tile[c][nl] = f[((size_t)bb * CB + c) * Nn + n0 + nl];
    }
    __syncthreads();
    for (int e = tid; e < CB * TILE_N; e += 256) {   // coalesced along c
        int nl = e >> 6, c = e & 63;
        featH[((size_t)bb * Nn + n0 + nl) * CB + c] = (_Float16)tile[c][nl];
    }
}

// ---------------------------------------------------------------------------
// Stage 2: G = relu(w2b @ relu(w2a @ feat + b2a) + b2b), f16 point-major
// ---------------------------------------------------------------------------
__global__ __launch_bounds__(256) void g_branch(
    const _Float16* __restrict__ featH,
    const _Float16* __restrict__ w2aH, const float* __restrict__ b2a,
    const _Float16* __restrict__ w2bH, const float* __restrict__ b2b,
    _Float16* __restrict__ GH, int Nn)
{
    __shared__ __align__(16) _Float16 xt[TILE_N * 64];    // [n][64]
    __shared__ __align__(16) _Float16 ht[TILE_N * 128];   // [n][128]
    const int blocksPerBatch = Nn / TILE_N;
    const int bb = blockIdx.x / blocksPerBatch;
    const int n0 = (blockIdx.x % blocksPerBatch) * TILE_N;
    const int tid = threadIdx.x, w = tid >> 5, lane = tid & 31;

    // Load 64x64 f16 activation tile (contiguous 8 KB) via async DMA to LDS.
    const _Float16* src = featH + ((size_t)bb * Nn + n0) * 64;
    for (int e = tid; e < TILE_N * 64 / 8; e += 256)
        async_copy16(src + e * 8, xt + e * 8);
    async_wait0();
    __syncthreads();

    // GEMM1: [128 x 64] x [64 x 64] -> ht
    gemm_stage_lds<2>(w2aH, b2a, /*Mtiles=*/8, /*Ktot=*/64, xt, ht, 128, w, lane);
    __syncthreads();

    // GEMM2: [64 x 128] x [128 x 64] -> GH (global f16 point-major)
    for (int j = w; j < 8; j += 8) {
        const int mb = j >> 2, nt = j & 3;
        v8f acc[2];
#pragma unroll
        for (int m = 0; m < 2; ++m) acc[m] = (v8f){0.f,0.f,0.f,0.f,0.f,0.f,0.f,0.f};
        wmma_block<2>(w2bH, 128, ht, mb * 32, nt * 16, lane, acc);
        const int n = n0 + nt * 16 + (lane & 15);
        _Float16* dst = GH + ((size_t)bb * Nn + n) * 64;
#pragma unroll
        for (int m = 0; m < 2; ++m) {
            const int mo = mb * 32 + m * 16 + ((lane & 16) ? 8 : 0);
#pragma unroll
            for (int vi = 0; vi < 8; ++vi) {
                float val = acc[m][vi] + b2b[mo + vi];
                dst[mo + vi] = (_Float16)(val > 0.f ? val : 0.f);
            }
        }
    }
}

// ---------------------------------------------------------------------------
// Stage 3: fused gather + layer1 MLP + (Gg + L) + layer3 MLP -> out f32
// ---------------------------------------------------------------------------
__global__ __launch_bounds__(256) void pointconv_main(
    const int* __restrict__ idxG,
    const _Float16* __restrict__ featH,
    const _Float16* __restrict__ GH,
    const _Float16* __restrict__ w1aH, const float* __restrict__ b1a,
    const _Float16* __restrict__ w1bH, const float* __restrict__ b1b,
    const _Float16* __restrict__ w3aH, const float* __restrict__ b3a,
    const _Float16* __restrict__ w3bH, const float* __restrict__ b3b,
    float* __restrict__ out, int Nn)
{
    __shared__ __align__(16) _Float16 bufA[TILE_N * 512];  // gf -> L+Gg
    __shared__ __align__(16) _Float16 bufB[TILE_N * 512];  // h1 -> t3
    __shared__ int sIdx[TILE_N * KN];
    const int blocksPerBatch = Nn / TILE_N;
    const int bb = blockIdx.x / blocksPerBatch;
    const int n0 = (blockIdx.x % blocksPerBatch) * TILE_N;
    const int tid = threadIdx.x, w = tid >> 5, lane = tid & 31;

    // 1. neighbor indices for this tile
    const int* ip = idxG + ((size_t)bb * Nn + n0) * KN;
    for (int e = tid; e < TILE_N * KN; e += 256) sIdx[e] = ip[e];
    __syncthreads();

    // 2. gather gf[n][k*64+c]: 2048 x 32B neighbor-row slices, async DMA to LDS
    for (int t = tid; t < TILE_N * KN * 4; t += 256) {
        int part = t & 3, kk = (t >> 2) & 7, nl = t >> 5;
        int j = sIdx[nl * KN + kk];
        async_copy32(featH + (((size_t)bb * Nn + j) << 6) + (part << 4),
                     bufA + nl * 512 + kk * 64 + (part << 4));
    }
    async_wait0();
    __syncthreads();

    // 3. h1 = relu(w1a @ gf + b1a): [512x512]x[512x64] -> bufB
    gemm_stage_lds<4>(w1aH, b1a, 32, 512, bufA, bufB, 512, w, lane);
    __syncthreads();

    // 4. L = relu(w1b @ h1 + b1b): [512x512]x[512x64] -> bufA
    gemm_stage_lds<4>(w1bH, b1b, 32, 512, bufB, bufA, 512, w, lane);
    __syncthreads();

    // 5. bufA += gathered G (packed f16 adds)
    for (int t = tid; t < TILE_N * KN * 4; t += 256) {
        int part = t & 3, kk = (t >> 2) & 7, nl = t >> 5;
        int j = sIdx[nl * KN + kk];
        const v8h* s = (const v8h*)(GH + (((size_t)bb * Nn + j) << 6) + (part << 4));
        v8h* d = (v8h*)(bufA + nl * 512 + kk * 64 + (part << 4));
        d[0] = d[0] + s[0];
        d[1] = d[1] + s[1];
    }
    __syncthreads();

    // 6. t3 = relu(w3a @ (Gg+L) + b3a): [256x512]x[512x64] -> bufB[:64*256]
    gemm_stage_lds<4>(w3aH, b3a, 16, 512, bufA, bufB, 256, w, lane);
    __syncthreads();

    // 7. out = relu(w3b @ t3 + b3b): [128x256]x[256x64] -> global f32 [B,128,N]
    for (int j = w; j < 8; j += 8) {
        const int mb = j >> 2, nt = j & 3;
        v8f acc[4];
#pragma unroll
        for (int m = 0; m < 4; ++m) acc[m] = (v8f){0.f,0.f,0.f,0.f,0.f,0.f,0.f,0.f};
        wmma_block<4>(w3bH, 256, bufB, mb * 64, nt * 16, lane, acc);
        const int n = n0 + nt * 16 + (lane & 15);
#pragma unroll
        for (int m = 0; m < 4; ++m) {
            const int mo = mb * 64 + m * 16 + ((lane & 16) ? 8 : 0);
#pragma unroll
            for (int vi = 0; vi < 8; ++vi) {
                const int o = mo + vi;
                float val = acc[m][vi] + b3b[o];
                out[((size_t)bb * 128 + o) * Nn + n] = val > 0.f ? val : 0.f;
            }
        }
    }
}

// ---------------------------------------------------------------------------
// Host launcher
// ---------------------------------------------------------------------------
extern "C" void kernel_launch(void* const* d_in, const int* in_sizes, int n_in,
                              void* d_out, int out_size, void* d_ws, size_t ws_size,
                              hipStream_t stream) {
    const float* feature = (const float*)d_in[0];
    const int*   idx     = (const int*)d_in[1];
    const float* w1a = (const float*)d_in[2];  const float* b1a = (const float*)d_in[3];
    const float* w1b = (const float*)d_in[4];  const float* b1b = (const float*)d_in[5];
    const float* w2a = (const float*)d_in[6];  const float* b2a = (const float*)d_in[7];
    const float* w2b = (const float*)d_in[8];  const float* b2b = (const float*)d_in[9];
    const float* w3a = (const float*)d_in[10]; const float* b3a = (const float*)d_in[11];
    const float* w3b = (const float*)d_in[12]; const float* b3b = (const float*)d_in[13];
    float* out = (float*)d_out;

    // Workspace layout (f16 elements):
    _Float16* base = (_Float16*)d_ws;
    size_t off = 0;
    _Float16* w1aH = base + off; off += (size_t)512 * 512;
    _Float16* w1bH = base + off; off += (size_t)512 * 512;
    _Float16* w2aH = base + off; off += (size_t)128 * 64;
    _Float16* w2bH = base + off; off += (size_t)64 * 128;
    _Float16* w3aH = base + off; off += (size_t)256 * 512;
    _Float16* w3bH = base + off; off += (size_t)128 * 256;
    _Float16* featH = base + off; off += (size_t)NB * NP * CB;
    _Float16* GH    = base + off; off += (size_t)NB * NP * 64;
    (void)ws_size; (void)n_in; (void)in_sizes; (void)out_size;

    auto cvt = [&](const float* s, _Float16* d, int n) {
        cvt_f32_f16<<<(n + 255) / 256, 256, 0, stream>>>(s, d, n);
    };
    cvt(w1a, w1aH, 512 * 512);
    cvt(w1b, w1bH, 512 * 512);
    cvt(w2a, w2aH, 128 * 64);
    cvt(w2b, w2bH, 64 * 128);
    cvt(w3a, w3aH, 256 * 512);
    cvt(w3b, w3bH, 128 * 256);

    const int blocks = NB * (NP / TILE_N);  // 2048
    feat_transpose<<<blocks, 256, 0, stream>>>(feature, featH, NP);
    g_branch<<<blocks, 256, 0, stream>>>(featH, w2aH, b2a, w2bH, b2b, GH, NP);
    pointconv_main<<<blocks, 256, 0, stream>>>(
        idx, featH, GH,
        w1aH, b1a, w1bH, b1b, w3aH, b3a, w3bH, b3b,
        out, NP);
}
